// DotProductAttention_57561151701313
// MI455X (gfx1250) — matrix operand order repaired and checked by reading, not compiled, not run on hardware
//
#include <hip/hip_runtime.h>
#include <hip/hip_bf16.h>
#include <math.h>

// ---------------- problem constants (from reference) ----------------
#define BB   2
#define HH   16
#define SEQ  2048
#define DK   64

typedef _Float16 half8  __attribute__((ext_vector_type(8)));
typedef _Float16 half16 __attribute__((ext_vector_type(16)));
typedef float    f32x8  __attribute__((ext_vector_type(8)));

constexpr int KV_TILE = 64;               // keys per inner iteration
constexpr int WAVES   = 4;                // waves per workgroup
constexpr int BLOCK_Q = WAVES * 16;       // 64 query rows per workgroup
constexpr int NITER   = SEQ / KV_TILE;    // 32 KV tiles

// padded LDS strides (bank-conflict avoidance: 72*2B rows -> row*36 mod 64
// hits distinct multiples of 4 for all 16 rows; 68 floats for score rows)
constexpr int KS = 72;                    // Klds row stride (f16)
constexpr int VS = 72;                    // Vt row stride (f16)
constexpr int SS = 68;                    // Sl row stride (f32)
constexpr int PS = 72;                    // Pl row stride (f16)

// softmax(score * scale) computed in exp2 units: t = score * scale * log2(e)
constexpr float SCALE_LOG2E = 0.125f * 1.44269504088896340736f; // 1/sqrt(64)*log2e

// wave-internal LDS exchange: wait for this wave's outstanding DS ops
__device__ __forceinline__ void lds_fence() {
  asm volatile("s_wait_dscnt 0" ::: "memory");
}

// A-matrix 16x32 f16 fragment (ISA 7.12.2): lane-half h holds
// K = h*8 + 0..7 (elements 0..7) and 16 + h*8 + 0..7 (elements 8..15).
__device__ __forceinline__ half16 load_fragA_lds(const _Float16* rowbase, int halfid) {
  const _Float16* p = rowbase + halfid * 8;
  half8 lo = *reinterpret_cast<const half8*>(p);
  half8 hi = *reinterpret_cast<const half8*>(p + 16);
  half16 r;
#pragma unroll
  for (int i = 0; i < 8; ++i) { r[i] = lo[i]; r[i + 8] = hi[i]; }
  return r;
}

// B-matrix 32x16 f16 fragment: lane = column n, lane-half h holds
// K = h*8 + 0..7 (elements 0..7) and 16 + h*8 + 0..7 (elements 8..15).
__device__ __forceinline__ half16 load_fragB_lds(const _Float16* rowbase, int halfid) {
  const _Float16* p = rowbase + halfid * 8;
  half8 lo = *reinterpret_cast<const half8*>(p);
  half8 hi = *reinterpret_cast<const half8*>(p + 16);
  half16 r;
#pragma unroll
  for (int i = 0; i < 8; ++i) { r[i] = lo[i]; r[i + 8] = hi[i]; }
  return r;
}

__global__ __launch_bounds__(WAVES * 32)
void DotProductAttention_fa_kernel(const float* __restrict__ Q,
                                   const float* __restrict__ K,
                                   const float* __restrict__ V,
                                   const int*   __restrict__ Msk,
                                   float*       __restrict__ O) {
  // ---------------- LDS ----------------
  __shared__ __attribute__((aligned(16))) _Float16 Klds[KV_TILE * KS];   // [key][feat], padded
  __shared__ __attribute__((aligned(16))) _Float16 Vt  [DK * VS];        // [feat][key], padded
  __shared__ __attribute__((aligned(16))) float    Sl  [WAVES][16 * SS];
  __shared__ __attribute__((aligned(16))) _Float16 Pl  [WAVES][16 * PS];
  __shared__ __attribute__((aligned(16))) float    Al  [WAVES][16];
  __shared__ __attribute__((aligned(16))) float    Ll  [WAVES][16];

  const int bh     = blockIdx.y;         // fused batch*head
  const int b      = bh / HH;
  const int q0     = blockIdx.x * BLOCK_Q;
  const int tid    = threadIdx.x;
  const int wave   = tid >> 5;
  const int lane   = tid & 31;
  const int halfid = lane >> 4;
  const int mrow   = lane & 15;

  const float* Qb = Q + (size_t)bh * SEQ * DK;
  const float* Kb = K + (size_t)bh * SEQ * DK;
  const float* Vb = V + (size_t)bh * SEQ * DK;
  const int*   Mb = Msk + (size_t)b * SEQ * SEQ;   // mask broadcasts over heads
  float*       Ob = O + (size_t)bh * SEQ * DK;

  const int qw0 = q0 + wave * 16;        // this wave's first query row

  // per-lane mask row base: row (qw0+mrow), column span [halfid*32, +32)
  const int* mask_base = Mb + (size_t)(qw0 + mrow) * SEQ + halfid * 32;

  // -------- load Q tile once as two f16 A-fragments (features 0..31, 32..63)
  half16 qa0, qa1;
  {
    const float* qrow = Qb + (size_t)(qw0 + mrow) * DK;
#pragma unroll
    for (int i = 0; i < 16; ++i) {
      int k = halfid * 8 + (i & 7) + ((i & 8) ? 16 : 0);
      qa0[i] = (_Float16)qrow[k];
      qa1[i] = (_Float16)qrow[32 + k];
    }
  }

  f32x8 acc[4] = {f32x8{}, f32x8{}, f32x8{}, f32x8{}};   // 16x64 output tile
  // per-row softmax state, duplicated on the two lanes owning row `mrow`
  float m_i = -INFINITY, l_i = 0.0f;

  // cooperative staging: 128 threads x 32 elems = 64x64 tile
  const int srow = tid >> 1;             // 0..63 (key row)
  const int sf0  = (tid & 1) * 32;       // feature start: 0 or 32

  for (int kt = 0; kt < NITER; ++kt) {
    const int kv0 = kt * KV_TILE;

    __syncthreads();                     // previous K/V tile fully consumed
    {
      const float4* kg = reinterpret_cast<const float4*>(Kb + (size_t)(kv0 + srow) * DK + sf0);
      const float4* vg = reinterpret_cast<const float4*>(Vb + (size_t)(kv0 + srow) * DK + sf0);
#pragma unroll
      for (int j = 0; j < 8; ++j) {
        float4 kk = kg[j];
        float4 vv = vg[j];
        int c0 = sf0 + j * 4;
        Klds[srow * KS + c0 + 0] = (_Float16)kk.x;
        Klds[srow * KS + c0 + 1] = (_Float16)kk.y;
        Klds[srow * KS + c0 + 2] = (_Float16)kk.z;
        Klds[srow * KS + c0 + 3] = (_Float16)kk.w;
        Vt[(c0 + 0) * VS + srow] = (_Float16)vv.x;
        Vt[(c0 + 1) * VS + srow] = (_Float16)vv.y;
        Vt[(c0 + 2) * VS + srow] = (_Float16)vv.z;
        Vt[(c0 + 3) * VS + srow] = (_Float16)vv.w;
      }
      if (kt + 1 < NITER) {              // prefetch next tile into caches
        __builtin_prefetch(Kb + (size_t)(kv0 + KV_TILE + srow) * DK + sf0, 0, 0);
        __builtin_prefetch(Vb + (size_t)(kv0 + KV_TILE + srow) * DK + sf0, 0, 0);
      }
    }
    __syncthreads();                     // tile staged

    // -------- S = Q K^T : preload all 8 B fragments, then 8 WMMAs
    // interleaved over 4 independent accumulators (deep XDL pipeline)
    f32x8 d[4] = {f32x8{}, f32x8{}, f32x8{}, f32x8{}};
    {
      half16 bk[2][4];
#pragma unroll
      for (int nt = 0; nt < 4; ++nt) {
        bk[0][nt] = load_fragB_lds(&Klds[(nt * 16 + mrow) * KS + 0],  halfid);
        bk[1][nt] = load_fragB_lds(&Klds[(nt * 16 + mrow) * KS + 32], halfid);
      }
#pragma unroll
      for (int nt = 0; nt < 4; ++nt)
        d[nt] = __builtin_amdgcn_wmma_f32_16x16x32_f16(false, qa0, false, bk[0][nt], (short)0, d[nt], false, false);
#pragma unroll
      for (int nt = 0; nt < 4; ++nt)
        d[nt] = __builtin_amdgcn_wmma_f32_16x16x32_f16(false, qa1, false, bk[1][nt], (short)0, d[nt], false, false);
    }

    // -------- scale (exp2 units) and spill scores to per-wave LDS
    // D layout: lane l, reg r -> row m = r + 8*(l>>4), col n = l&15
    // (mask is applied later in the softmax pass with vectorized loads)
#pragma unroll
    for (int r = 0; r < 8; ++r) {
      int m = r + 8 * halfid;
#pragma unroll
      for (int nt = 0; nt < 4; ++nt)
        Sl[wave][m * SS + nt * 16 + mrow] = d[nt][r] * SCALE_LOG2E;
    }
    lds_fence();

    // -------- online softmax: rows owned by lane pairs (mrow, lane^16)
    {
      // vectorized mask load: 8 x int4 covering this lane's 32 columns
      const int4* mp = reinterpret_cast<const int4*>(mask_base + kv0);
      int mk[32];
#pragma unroll
      for (int j = 0; j < 8; ++j) {
        int4 m4 = mp[j];
        mk[j * 4 + 0] = m4.x; mk[j * 4 + 1] = m4.y;
        mk[j * 4 + 2] = m4.z; mk[j * 4 + 3] = m4.w;
      }

      const float* srowp = &Sl[wave][mrow * SS + halfid * 32];
      float sv[32];
#pragma unroll
      for (int c = 0; c < 32; ++c) sv[c] = srowp[c];
#pragma unroll
      for (int c = 0; c < 32; ++c) sv[c] = (mk[c] == 0) ? -INFINITY : sv[c];

      float pm = -INFINITY;
#pragma unroll
      for (int c = 0; c < 32; ++c) pm = fmaxf(pm, sv[c]);
      float tmax = fmaxf(pm, __shfl_xor(pm, 16, 32));     // cross-half combine
      float mn   = fmaxf(m_i, tmax);
      float alpha = (m_i > -INFINITY) ? __builtin_amdgcn_exp2f(m_i - mn) : 0.0f;

      _Float16* prow = &Pl[wave][mrow * PS + halfid * 32];
      float ps = 0.0f;
#pragma unroll
      for (int c = 0; c < 32; ++c) {
        float sc = sv[c];
        float p  = (sc > -INFINITY) ? __builtin_amdgcn_exp2f(sc - mn) : 0.0f;
        ps += p;
        prow[c] = (_Float16)p;
      }
      float sum = ps + __shfl_xor(ps, 16, 32);            // cross-half combine
      l_i = l_i * alpha + sum;
      m_i = mn;
      Al[wave][mrow] = alpha;   // both halves write identical value (benign)
    }
    lds_fence();

    // -------- rescale accumulators by per-row alpha (vector LDS read)
    {
      const float4* ap = reinterpret_cast<const float4*>(&Al[wave][halfid * 8]);
      float4 a0 = ap[0], a1 = ap[1];
      float av[8] = {a0.x, a0.y, a0.z, a0.w, a1.x, a1.y, a1.z, a1.w};
#pragma unroll
      for (int r = 0; r < 8; ++r) {
        float a = av[r];
        acc[0][r] *= a; acc[1][r] *= a; acc[2][r] *= a; acc[3][r] *= a;
      }
    }

    // -------- O += P * V : preload fragments, then 8 WMMAs interleaved
    {
      half16 pa0 = load_fragA_lds(&Pl[wave][mrow * PS + 0],  halfid);
      half16 pa1 = load_fragA_lds(&Pl[wave][mrow * PS + 32], halfid);
      half16 bv[2][4];
#pragma unroll
      for (int nt = 0; nt < 4; ++nt) {
        bv[0][nt] = load_fragB_lds(&Vt[(nt * 16 + mrow) * VS + 0],  halfid);
        bv[1][nt] = load_fragB_lds(&Vt[(nt * 16 + mrow) * VS + 32], halfid);
      }
#pragma unroll
      for (int nt = 0; nt < 4; ++nt)
        acc[nt] = __builtin_amdgcn_wmma_f32_16x16x32_f16(false, pa0, false, bv[0][nt], (short)0, acc[nt], false, false);
#pragma unroll
      for (int nt = 0; nt < 4; ++nt)
        acc[nt] = __builtin_amdgcn_wmma_f32_16x16x32_f16(false, pa1, false, bv[1][nt], (short)0, acc[nt], false, false);
    }
  }

  // -------- epilogue: out = acc / l  (owner lanes publish 1/l once)
  Ll[wave][mrow] = 1.0f / l_i;           // duplicated identical write
  lds_fence();
  {
    const float4* lp = reinterpret_cast<const float4*>(&Ll[wave][halfid * 8]);
    float4 l0 = lp[0], l1 = lp[1];
    float lv[8] = {l0.x, l0.y, l0.z, l0.w, l1.x, l1.y, l1.z, l1.w};
#pragma unroll
    for (int r = 0; r < 8; ++r) {
      int m = r + 8 * halfid;
      float inv = lv[r];
      float* orow = Ob + (size_t)(qw0 + m) * DK;
      orow[ 0 + mrow] = acc[0][r] * inv;
      orow[16 + mrow] = acc[1][r] * inv;
      orow[32 + mrow] = acc[2][r] * inv;
      orow[48 + mrow] = acc[3][r] * inv;
    }
  }
}

extern "C" void kernel_launch(void* const* d_in, const int* in_sizes, int n_in,
                              void* d_out, int out_size, void* d_ws, size_t ws_size,
                              hipStream_t stream) {
  (void)in_sizes; (void)n_in; (void)out_size; (void)d_ws; (void)ws_size;
  const float* q = (const float*)d_in[0];
  const float* k = (const float*)d_in[1];
  const float* v = (const float*)d_in[2];
  const int*   m = (const int*)  d_in[3];
  float*       o = (float*)d_out;

  dim3 grid(SEQ / BLOCK_Q, BB * HH);   // (32, 32)
  dim3 block(WAVES * 32);              // 128 threads = 4 waves (wave32)
  DotProductAttention_fa_kernel<<<grid, block, 0, stream>>>(q, k, v, m, o);
}
